// MultiQueryAttention_15822659519062
// MI455X (gfx1250) — matrix operand order repaired and checked by reading, not compiled, not run on hardware
//
#include <hip/hip_runtime.h>

typedef __bf16 bf16_t;
typedef __bf16 v8bf  __attribute__((ext_vector_type(8)));
typedef __bf16 v16bf __attribute__((ext_vector_type(16)));
typedef float  v8f   __attribute__((ext_vector_type(8)));

#define D_MODEL 2048
#define N_HEADS 16
#define D_HEAD  128
#define BATCH   2
#define SEQ     2048
#define BS      (BATCH*SEQ)   // 4096

// ---------------------------------------------------------------------------
// Elementwise convert fp32 -> bf16
// ---------------------------------------------------------------------------
__global__ void cvt_f32_bf16(const float* __restrict__ in, bf16_t* __restrict__ out, int n) {
    int i = blockIdx.x * blockDim.x + threadIdx.x;
    if (i < n) out[i] = (bf16_t)in[i];
}

// Transpose fp32 (K x N) -> bf16 (N x K)
__global__ void transpose_f32_bf16(const float* __restrict__ in, bf16_t* __restrict__ out,
                                   int K, int N) {
    int i = blockIdx.x * blockDim.x + threadIdx.x;   // over N*K
    if (i >= N * K) return;
    int n = i / K, k = i % K;
    out[i] = (bf16_t)in[(size_t)k * N + n];
}

// Transpose V: (B, S, Dh) bf16 -> (B, Dh, S) bf16
__global__ void transpose_v(const bf16_t* __restrict__ v, bf16_t* __restrict__ vt) {
    int i = blockIdx.x * blockDim.x + threadIdx.x;   // over B*Dh*S
    if (i >= BATCH * D_HEAD * SEQ) return;
    int b = i / (D_HEAD * SEQ);
    int r = i % (D_HEAD * SEQ);
    int d = r / SEQ, s = r % SEQ;
    vt[i] = v[(size_t)b * SEQ * D_HEAD + (size_t)s * D_HEAD + d];
}

// ---------------------------------------------------------------------------
// bf16 WMMA GEMM: C(M x N) = alpha * A(M x K, row major) * Bt(N x K)^T
// One wave computes a 64x64 tile. OUT_F32 selects fp32 vs bf16 output.
// ---------------------------------------------------------------------------
template<bool OUT_F32>
__global__ __launch_bounds__(128)
void gemm_bf16_wmma(const bf16_t* __restrict__ A, const bf16_t* __restrict__ Bt,
                    void* __restrict__ C, int M, int N, int K, float alpha) {
    const int lane = threadIdx.x & 31;
    const int wid  = blockIdx.x * (blockDim.x >> 5) + (threadIdx.x >> 5);
    const int ntiles = N >> 6;
    const int tm = (wid / ntiles) << 6;
    const int tn = (wid % ntiles) << 6;
    const int half = lane >> 4;    // 0 | 1
    const int l15  = lane & 15;

    v8f zero = {};
    v8f acc[4][4];
#pragma unroll
    for (int i = 0; i < 4; i++)
#pragma unroll
        for (int j = 0; j < 4; j++) acc[i][j] = zero;

    for (int kk = 0; kk < K; kk += 32) {
        v16bf a[4], b[4];
#pragma unroll
        for (int i = 0; i < 4; i++) {
            // A-fragment (16x32 bf16): lane holds row (lane&15),
            // K = half*16..+16 in elems 0..15
            const bf16_t* ap = A + (size_t)(tm + i * 16 + l15) * K + kk + half * 16;
            v8bf lo = *(const v8bf*)ap;
            v8bf hi = *(const v8bf*)(ap + 8);
#pragma unroll
            for (int e = 0; e < 8; e++) { a[i][e] = lo[e]; a[i][8 + e] = hi[e]; }
        }
#pragma unroll
        for (int j = 0; j < 4; j++) {
            // B-fragment (32x16 bf16): lane holds col (lane&15), K = half*16..+16
            const bf16_t* bp = Bt + (size_t)(tn + j * 16 + l15) * K + kk + half * 16;
            b[j] = *(const v16bf*)bp;
        }
#pragma unroll
        for (int i = 0; i < 4; i++)
#pragma unroll
            for (int j = 0; j < 4; j++)
                acc[i][j] = __builtin_amdgcn_wmma_f32_16x16x32_bf16(
                    false, a[i], false, b[j], (short)0, acc[i][j], false, false);
    }

    // C/D layout: lane holds col (lane&15); VGPR r holds row r + half*8
#pragma unroll
    for (int i = 0; i < 4; i++) {
#pragma unroll
        for (int j = 0; j < 4; j++) {
            int col = tn + j * 16 + l15;
#pragma unroll
            for (int r = 0; r < 8; r++) {
                int row = tm + i * 16 + r + half * 8;
                float v = acc[i][j][r] * alpha;
                if (OUT_F32)
                    ((float*)C)[(size_t)row * N + col] = v;
                else
                    ((bf16_t*)C)[(size_t)row * N + col] = (bf16_t)v;
            }
        }
    }
}

// ---------------------------------------------------------------------------
// Flash MQA: one wave handles 16 query rows of one (batch, head).
// Q:  (B, S, D_MODEL) bf16, PRE-SCALED by 1/sqrt(D_HEAD)
// Km: (B, S, 128) bf16     (shared single KV head)
// Vt: (B, 128, S) bf16     (V transposed for contiguous B-fragments)
// Ctx out: (B, S, D_MODEL) bf16
// 64 keys per iteration; row-sum kept lane-partial (reduced once at the end).
// ---------------------------------------------------------------------------
__global__ __launch_bounds__(128)
void mqa_flash(const bf16_t* __restrict__ Q, const bf16_t* __restrict__ Km,
               const bf16_t* __restrict__ Vt, bf16_t* __restrict__ Ctx) {
    __shared__ __align__(16) bf16_t Pst[4][16 * 64];

    const int lane = threadIdx.x & 31;
    const int w    = threadIdx.x >> 5;
    const int wid  = blockIdx.x * 4 + w;
    const int qtiles = SEQ / 16;                 // 128
    const int qt = wid % qtiles;
    const int h  = (wid / qtiles) % N_HEADS;
    const int b  = wid / (qtiles * N_HEADS);
    const int half = lane >> 4;
    const int l15  = lane & 15;

    const bf16_t* Qb = Q  + (size_t)b * SEQ * D_MODEL;
    const bf16_t* Kb = Km + (size_t)b * SEQ * D_HEAD;
    const bf16_t* Vb = Vt + (size_t)b * D_HEAD * SEQ;

    // Resident Q fragments: 16 rows x 128 d = 4 K-steps of 32
    v16bf qf[4];
    {
        const bf16_t* qp = Qb + (size_t)(qt * 16 + l15) * D_MODEL + h * D_HEAD;
#pragma unroll
        for (int dk = 0; dk < 4; dk++) {
            const bf16_t* p = qp + dk * 32 + half * 16;
            v8bf lo = *(const v8bf*)p;
            v8bf hi = *(const v8bf*)(p + 8);
#pragma unroll
            for (int e = 0; e < 8; e++) { qf[dk][e] = lo[e]; qf[dk][8 + e] = hi[e]; }
        }
    }

    v8f zero = {};
    v8f o[8];                                    // 16 rows x 128 cols accumulator
#pragma unroll
    for (int t = 0; t < 8; t++) o[t] = zero;
    float rmax[8], rsum[8];                      // rsum is LANE-PARTIAL
#pragma unroll
    for (int r = 0; r < 8; r++) { rmax[r] = -1e30f; rsum[r] = 0.0f; }

    for (int kb = 0; kb < SEQ; kb += 64) {
        // ---- scores: four 16-key tiles, K-dim = d (128) ----
        v8f s[4];
#pragma unroll
        for (int j = 0; j < 4; j++) s[j] = zero;
#pragma unroll
        for (int dk = 0; dk < 4; dk++) {
            int doff = dk * 32 + half * 16;
#pragma unroll
            for (int j = 0; j < 4; j++) {
                v16bf kf = *(const v16bf*)(Kb + (size_t)(kb + j * 16 + l15) * D_HEAD + doff);
                s[j] = __builtin_amdgcn_wmma_f32_16x16x32_bf16(false, qf[dk], false, kf,
                                                               (short)0, s[j], false, false);
            }
        }

        // prefetch next chunk of K rows / V columns into cache
        if (kb + 64 < SEQ) {
            __builtin_prefetch(Kb + (size_t)(kb + 64 + lane) * D_HEAD, 0, 0);
            __builtin_prefetch(Kb + (size_t)(kb + 96 + lane) * D_HEAD, 0, 0);
            __builtin_prefetch(Vb + (size_t)(lane * 4) * SEQ + kb + 64, 0, 0);
        }

        // ---- online softmax: only the MAX needs a cross-lane reduction ----
        float c[8];
#pragma unroll
        for (int r = 0; r < 8; r++) {
            float mx = fmaxf(fmaxf(s[0][r], s[1][r]), fmaxf(s[2][r], s[3][r]));
#pragma unroll
            for (int off = 1; off < 16; off <<= 1)
                mx = fmaxf(mx, __shfl_xor(mx, off, 32));
            float mnew = fmaxf(rmax[r], mx);
            c[r] = __expf(rmax[r] - mnew);
            rmax[r] = mnew;
            float ps = 0.0f;
#pragma unroll
            for (int j = 0; j < 4; j++) {
                float p = __expf(s[j][r] - mnew);
                s[j][r] = p;
                ps += p;
            }
            rsum[r] = rsum[r] * c[r] + ps;       // lane-partial row sum
        }
#pragma unroll
        for (int t = 0; t < 8; t++)
#pragma unroll
            for (int r = 0; r < 8; r++) o[t][r] *= c[r];

        // ---- transpose P (C-layout -> A-layout) through per-wave LDS ----
        bf16_t* pst = Pst[w];
#pragma unroll
        for (int r = 0; r < 8; r++) {
            int m = r + half * 8;
#pragma unroll
            for (int j = 0; j < 4; j++)
                pst[m * 64 + j * 16 + l15] = (bf16_t)s[j][r];
        }
        asm volatile("s_wait_dscnt 0" ::: "memory");   // same-wave LDS RAW fence
        v16bf pf[2];
#pragma unroll
        for (int kc = 0; kc < 2; kc++) {
            const bf16_t* pp = pst + l15 * 64 + kc * 32 + half * 16;
            v8bf lo = *(const v8bf*)pp;
            v8bf hi = *(const v8bf*)(pp + 8);
#pragma unroll
            for (int e = 0; e < 8; e++) { pf[kc][e] = lo[e]; pf[kc][8 + e] = hi[e]; }
        }

        // ---- context += P(16x64) @ V(64x128): 8 col tiles x 2 K-steps ----
#pragma unroll
        for (int t = 0; t < 8; t++) {
#pragma unroll
            for (int kc = 0; kc < 2; kc++) {
                v16bf vf = *(const v16bf*)(Vb + (size_t)(t * 16 + l15) * SEQ
                                           + kb + kc * 32 + half * 16);
                o[t] = __builtin_amdgcn_wmma_f32_16x16x32_bf16(false, pf[kc], false, vf,
                                                               (short)0, o[t], false, false);
            }
        }
    }

    // ---- final cross-lane row-sum reduction (once), normalize, store ----
    float inv[8];
#pragma unroll
    for (int r = 0; r < 8; r++) {
        float ps = rsum[r];
#pragma unroll
        for (int off = 1; off < 16; off <<= 1)
            ps += __shfl_xor(ps, off, 32);
        inv[r] = 1.0f / ps;
    }
#pragma unroll
    for (int t = 0; t < 8; t++) {
        int col = h * D_HEAD + t * 16 + l15;
#pragma unroll
        for (int r = 0; r < 8; r++) {
            int row = qt * 16 + r + half * 8;
            Ctx[((size_t)b * SEQ + row) * D_MODEL + col] = (bf16_t)(o[t][r] * inv[r]);
        }
    }
}

// ---------------------------------------------------------------------------
// Host-side orchestration
// ---------------------------------------------------------------------------
extern "C" void kernel_launch(void* const* d_in, const int* in_sizes, int n_in,
                              void* d_out, int out_size, void* d_ws, size_t ws_size,
                              hipStream_t stream) {
    const float* x  = (const float*)d_in[0];
    const float* Wq = (const float*)d_in[1];
    const float* Wk = (const float*)d_in[2];
    const float* Wv = (const float*)d_in[3];
    const float* Wo = (const float*)d_in[4];
    float* out = (float*)d_out;

    char* ws = (char*)d_ws;
    // workspace layout (bytes)
    bf16_t* xb  = (bf16_t*)(ws + 0);           // 16 MB : x bf16 (BS x D)
    bf16_t* WqT = (bf16_t*)(ws + 16777216);    //  8 MB : W_q^T (D x D)
    bf16_t* WkT = (bf16_t*)(ws + 25165824);    // .5 MB : W_k^T (128 x D)
    bf16_t* WvT = (bf16_t*)(ws + 25690112);    // .5 MB : W_v^T (128 x D)
    bf16_t* WoT = (bf16_t*)(ws + 26214400);    //  8 MB : W_o^T (D x D)
    bf16_t* Qb  = (bf16_t*)(ws + 34603008);    // 16 MB : Q (BS x D), pre-scaled
    bf16_t* Kb  = (bf16_t*)(ws + 51380224);    //  1 MB : K (BS x 128)
    bf16_t* Vb  = (bf16_t*)(ws + 52428800);    //  1 MB : V (BS x 128)
    bf16_t* VTb = (bf16_t*)(ws + 53477376);    //  1 MB : V^T (B x 128 x S)
    bf16_t* Ctx = (bf16_t*)(ws + 54525952);    // 16 MB : context (BS x D)

    const float qscale = 0.08838834764831845f; // 1/sqrt(128), folded into Q

    // 1) convert / transpose to bf16
    {
        int n = BS * D_MODEL;
        cvt_f32_bf16<<<(n + 255) / 256, 256, 0, stream>>>(x, xb, n);
    }
    transpose_f32_bf16<<<(D_MODEL * D_MODEL + 255) / 256, 256, 0, stream>>>(Wq, WqT, D_MODEL, D_MODEL);
    transpose_f32_bf16<<<(D_MODEL * D_HEAD + 255) / 256, 256, 0, stream>>>(Wk, WkT, D_MODEL, D_HEAD);
    transpose_f32_bf16<<<(D_MODEL * D_HEAD + 255) / 256, 256, 0, stream>>>(Wv, WvT, D_MODEL, D_HEAD);
    transpose_f32_bf16<<<(D_MODEL * D_MODEL + 255) / 256, 256, 0, stream>>>(Wo, WoT, D_MODEL, D_MODEL);

    // 2) projections (bf16 WMMA GEMMs); Q gets the softmax scale folded in
    gemm_bf16_wmma<false><<<(BS / 64) * (D_MODEL / 64) / 4, 128, 0, stream>>>(xb, WqT, Qb, BS, D_MODEL, D_MODEL, qscale);
    gemm_bf16_wmma<false><<<(BS / 64) * (D_HEAD / 64) / 4, 128, 0, stream>>>(xb, WkT, Kb, BS, D_HEAD, D_MODEL, 1.0f);
    gemm_bf16_wmma<false><<<(BS / 64) * (D_HEAD / 64) / 4, 128, 0, stream>>>(xb, WvT, Vb, BS, D_HEAD, D_MODEL, 1.0f);
    transpose_v<<<(BATCH * D_HEAD * SEQ + 255) / 256, 256, 0, stream>>>(Vb, VTb);

    // 3) flash attention (one wave per 16 query rows per (b,h))
    mqa_flash<<<BATCH * N_HEADS * (SEQ / 16) / 4, 128, 0, stream>>>(Qb, Kb, VTb, Ctx);

    // 4) output projection (fp32 out)
    gemm_bf16_wmma<true><<<(BS / 64) * (D_MODEL / 64) / 4, 128, 0, stream>>>(Ctx, WoT, out, BS, D_MODEL, D_MODEL, 1.0f);
}
